// GNNInteractionNetwork_9612136808838
// MI455X (gfx1250) — matrix operand order repaired and checked by reading, not compiled, not run on hardware
//
#include <hip/hip_runtime.h>

typedef __attribute__((ext_vector_type(16))) _Float16 v16h;
typedef __attribute__((ext_vector_type(8)))  float    v8f;

#define HIDDEN 64
#define NPART 1024
#define ROWS_PER_BLOCK 8
#define THREADS (ROWS_PER_BLOCK * 32)

// 16-bit A-matrix 16x32 VGPR layout (CDNA5 ISA):
// VGPR v, lanes 0-15 hold K = 2v (+0|+8 group), lanes 16-31 the +8 shifted set.
__device__ __forceinline__ int a_kmap(int c, int halfid, int e) {
    int v = e >> 1;
    int koff = 2 * v + ((v < 4) ? 0 : 8);
    return 32 * c + 8 * halfid + koff + (e & 1);
}

__global__ __launch_bounds__(THREADS)
void gnn_edge_wmma(const float* __restrict__ pos,
                   const float* __restrict__ eps_p,
                   const float* __restrict__ w1,
                   const float* __restrict__ b1,
                   const float* __restrict__ W2,
                   const float* __restrict__ b2,
                   const float* __restrict__ w3,
                   const float* __restrict__ b3_p,
                   float* __restrict__ out)
{
    __shared__ float ldsPos[NPART * 3];                 // 12 KB: pos[b] staged once
    __shared__ float ldsM[ROWS_PER_BLOCK][16][17];      // padded: bank-conflict-free rows

    const int tid    = threadIdx.x;
    const int lane   = tid & 31;
    const int wave   = tid >> 5;
    const int halfid = lane >> 4;   // 0: lanes 0-15, 1: lanes 16-31
    const int col    = lane & 15;

    const int row  = blockIdx.x * ROWS_PER_BLOCK + wave; // flattened (b,i)
    const int bidx = row >> 10;
    const int i    = row & (NPART - 1);

    // cooperative stage of pos[bidx] into LDS
    const float* posb = pos + (size_t)bidx * NPART * 3;
    for (int idx = tid; idx < NPART * 3; idx += THREADS)
        ldsPos[idx] = posb[idx];
    __syncthreads();

    const float eps  = eps_p[0];
    const float eps2 = eps * eps;
    const float b3   = b3_p[0];

    // ---- hoist per-wave constant fragments into registers ----
    // layer-1 weights in A-fragment layout (f16)
    v16h w1h[2], b1h[2];
#pragma unroll
    for (int c = 0; c < 2; ++c)
#pragma unroll
        for (int e = 0; e < 16; ++e) {
            int k = a_kmap(c, halfid, e);
            w1h[c][e] = (_Float16)w1[k];
            b1h[c][e] = (_Float16)b1[k];
        }

    // W2 in B-fragment layout: B 32x16, lanes 0-15 hold K=0..7,16..23, lanes 16-31 K=8..15,24..31
    v16h Bf[4][2];
#pragma unroll
    for (int t = 0; t < 4; ++t)
#pragma unroll
        for (int c = 0; c < 2; ++c)
#pragma unroll
            for (int e = 0; e < 16; ++e) {
                int k = 32 * c + 8 * halfid + e + ((e >= 8) ? 8 : 0);
                Bf[t][c][e] = (_Float16)W2[k * HIDDEN + t * 16 + col];
            }

    float b2v[4], w3v[4];
#pragma unroll
    for (int t = 0; t < 4; ++t) {
        b2v[t] = b2[t * 16 + col];
        w3v[t] = w3[t * 16 + col];
    }

    const float pix = ldsPos[i * 3 + 0];
    const float piy = ldsPos[i * 3 + 1];
    const float piz = ldsPos[i * 3 + 2];

    float fx = 0.f, fy = 0.f, fz = 0.f;

    for (int j0 = 0; j0 < NPART; j0 += 16) {
        const int jm = j0 + col;  // this lane's pair row (A-matrix M = col)
        float dxx = ldsPos[jm * 3 + 0] - pix;
        float dxy = ldsPos[jm * 3 + 1] - piy;
        float dxz = ldsPos[jm * 3 + 2] - piz;
        float r2  = dxx * dxx + dxy * dxy + dxz * dxz;
        float r2s = r2 + ((jm == i) ? 1.0f : 0.0f);   // r2_safe (diagonal guard)

        // layer 1 in f16, written directly in A-fragment layout
        _Float16 r2hh = (_Float16)r2s;
        v16h a0, a1;
#pragma unroll
        for (int e = 0; e < 16; ++e) {
            _Float16 h0 = r2hh * w1h[0][e] + b1h[0][e];
            _Float16 h1 = r2hh * w1h[1][e] + b1h[1][e];
            a0[e] = (h0 > (_Float16)0.f) ? h0 : (_Float16)0.f;
            a1[e] = (h1 > (_Float16)0.f) ? h1 : (_Float16)0.f;
        }

        // layer 2: [16 pairs x 64] = A[16x64] * W2[64x64], 4 N-tiles x 2 chained K-chunks
        v8f accs[4];
        v8f zero = {};
#pragma unroll
        for (int t = 0; t < 4; ++t) {
            v8f acc = __builtin_amdgcn_wmma_f32_16x16x32_f16(
                false, a0, false, Bf[t][0], (short)0, zero, false, false);
            acc = __builtin_amdgcn_wmma_f32_16x16x32_f16(
                false, a1, false, Bf[t][1], (short)0, acc, false, false);
            accs[t] = acc;
        }

        // layer 3 partials: per lane, channel = t*16+col, row m = r + 8*halfid
#pragma unroll
        for (int r = 0; r < 8; ++r) {
            float p = 0.f;
#pragma unroll
            for (int t = 0; t < 4; ++t) {
                float h2 = accs[t][r] + b2v[t];
                h2 = (h2 > 0.f) ? h2 : 0.f;
                p = fmaf(h2, w3v[t], p);
            }
            ldsM[wave][r + 8 * halfid][col] = p;
        }
        __builtin_amdgcn_sched_barrier(0);
        // intra-wave LDS RAW: DS ops are in-order per wave, no workgroup barrier needed

        if (lane < 16) {  // lane m finishes pair j0+m (it already has dx/r2s for it)
            float msum = 0.f;
#pragma unroll
            for (int c16 = 0; c16 < 16; ++c16)
                msum += ldsM[wave][lane][c16];
            float mval = msum + b3;
            float tt  = r2s + eps2;
            float inv = __frsqrt_rn(tt);
            float mag = inv * inv * inv;                // (r2s + eps^2)^-1.5
            float wgt = (jm == i) ? 0.f : mval * mag;   // notdiag mask
            fx = fmaf(wgt, dxx, fx);
            fy = fmaf(wgt, dxy, fy);
            fz = fmaf(wgt, dxz, fz);
        }
        __builtin_amdgcn_sched_barrier(0);
    }

    // reduce force over the 16 pair-lanes (lanes 16-31 hold zeros)
#pragma unroll
    for (int off = 8; off >= 1; off >>= 1) {
        fx += __shfl_xor(fx, off, 32);
        fy += __shfl_xor(fy, off, 32);
        fz += __shfl_xor(fz, off, 32);
    }
    if (lane == 0) {
        out[row * 3 + 0] = fx;
        out[row * 3 + 1] = fy;
        out[row * 3 + 2] = fz;
    }
}

extern "C" void kernel_launch(void* const* d_in, const int* in_sizes, int n_in,
                              void* d_out, int out_size, void* d_ws, size_t ws_size,
                              hipStream_t stream) {
    const float* pos = (const float*)d_in[0];
    const float* eps = (const float*)d_in[1];
    const float* w1  = (const float*)d_in[2];
    const float* b1  = (const float*)d_in[3];
    const float* W2  = (const float*)d_in[4];
    const float* b2  = (const float*)d_in[5];
    const float* w3  = (const float*)d_in[6];
    const float* b3  = (const float*)d_in[7];
    float* out = (float*)d_out;

    const int Bn = in_sizes[0] / (NPART * 3);          // batch count (2)
    dim3 grid((Bn * NPART) / ROWS_PER_BLOCK);          // 256 blocks
    gnn_edge_wmma<<<grid, THREADS, 0, stream>>>(pos, eps, w1, b1, W2, b2, w3, b3, out);
}
